// CSRNetGuided_89300960019199
// MI455X (gfx1250) — matrix operand order repaired and checked by reading, not compiled, not run on hardware
//
#include <hip/hip_runtime.h>
#include <hip/hip_bf16.h>

// ---------------- types ----------------
typedef __attribute__((ext_vector_type(16))) __bf16 v16bf;
typedef __attribute__((ext_vector_type(8)))  __bf16 v8bf;
typedef __attribute__((ext_vector_type(8)))  float  v8f;
typedef __attribute__((ext_vector_type(4)))  unsigned int v4u;
typedef __attribute__((ext_vector_type(4)))  int v4i;
typedef __attribute__((ext_vector_type(8)))  int v8i;

__device__ __forceinline__ unsigned short f2bf(float f) {
    unsigned u = __float_as_uint(f);
    unsigned r = u + 0x7FFFu + ((u >> 16) & 1u);   // round-to-nearest-even
    return (unsigned short)(r >> 16);
}
__device__ __forceinline__ float bf2f(unsigned short h) {
    return __uint_as_float(((unsigned)h) << 16);
}

#define BATCH 8
#define PW 68            // padded (64 + 2*2)
#define NPIX 4096
#define BN_COUNT 32768.f // B*H*W

#if __has_builtin(__builtin_amdgcn_tensor_load_to_lds)
#define USE_TDM 1
#else
#define USE_TDM 0
#endif

#if USE_TDM
// 2-D TDM copy: rows x (tile0_8B*8 bytes), row stride stride0_8B*8 bytes, global -> LDS.
// D# per CDNA5 ISA ch.8: group0 {count=1, lds_addr, global_addr, type=2},
// group1 {data_size=3(8B), tensor_dim0/1, tile_dim0/1, dim0_stride}.
__device__ __forceinline__ void tdm_load_2d(unsigned lds_off, unsigned long long gaddr,
                                            unsigned tile0_8B, unsigned stride0_8B,
                                            unsigned rows) {
    v4u g0;
    g0.x = 1u;                                         // count=1, user mode
    g0.y = lds_off;                                    // LDS byte address
    g0.z = (unsigned)gaddr;                            // global_addr[31:0]
    g0.w = (unsigned)(gaddr >> 32) | (2u << 30);       // global_addr[56:32] | type=2
    v8i g1;
    g1[0] = (int)(3u << 16);                           // data_size = 8B
    g1[1] = (int)((tile0_8B & 0xFFFFu) << 16);         // tensor_dim0[15:0]
    g1[2] = (int)((tile0_8B >> 16) | ((rows & 0xFFFFu) << 16)); // td0[31:16] | td1[15:0]
    g1[3] = (int)((rows >> 16) | ((tile0_8B & 0xFFFFu) << 16)); // td1[31:16] | tile_dim0
    g1[4] = (int)rows;                                 // tile_dim1 | tile_dim2=0
    g1[5] = (int)stride0_8B;                           // tensor_dim0_stride[31:0]
    g1[6] = 0;
    g1[7] = 0;
    v4i gz = {0, 0, 0, 0};
#if defined(__clang_major__) && (__clang_major__ >= 23)
    v8i gz8 = {0, 0, 0, 0, 0, 0, 0, 0};
    __builtin_amdgcn_tensor_load_to_lds(g0, g1, gz, gz, gz8, 0);
#else
    __builtin_amdgcn_tensor_load_to_lds(g0, g1, gz, gz, 0);
#endif
    __builtin_amdgcn_s_wait_tensorcnt(0);
}
#endif

// ---------------- generic zero fill ----------------
__global__ void fill_zero_u32(unsigned* __restrict__ p, size_t n) {
    size_t t = (size_t)blockIdx.x * blockDim.x + threadIdx.x;
    if (t < n) p[t] = 0u;
}

// ---------------- pack x (NCHW f32) -> padded NHWC bf16 ----------------
__global__ void pack_input(const float* __restrict__ x, unsigned short* __restrict__ xp) {
    size_t t = (size_t)blockIdx.x * blockDim.x + threadIdx.x;
    if (t >= (size_t)BATCH * 512 * NPIX) return;
    int xw = (int)(t & 63);
    int yw = (int)((t >> 6) & 63);
    int c  = (int)((t >> 12) & 511);
    int b  = (int)(t >> 21);
    xp[((size_t)((b * PW + yw + 2) * PW + xw + 2)) * 512 + c] = f2bf(x[t]);
}

// ---------------- hypernet: wa[b][k] = y[b]·fc_w[k] + fc_b[k] ----------------
__global__ void hyper_wa(const float* __restrict__ y, const float* __restrict__ fcw,
                         const float* __restrict__ fcb, float* __restrict__ wa) {
    int t = blockIdx.x * blockDim.x + threadIdx.x;
    if (t >= BATCH * 100) return;
    int b = t / 100, k = t % 100;
    const float* wr = fcw + (size_t)k * 512;
    const float* yr = y + (size_t)b * 512;
    float s = 0.f;
    #pragma unroll 8
    for (int j = 0; j < 512; ++j) s += yr[j] * wr[j];
    wa[t] = s + fcb[k];
}

// ---------------- weight-gen stage 1: A[b][o][t] = WB[o,t]*wa[b,t], K padded to 128
template <int COUT>
__global__ void wscale_kernel(const float* __restrict__ WB, const float* __restrict__ wa,
                              unsigned short* __restrict__ A) {
    constexpr int M3 = 3 * COUT;
    size_t t = (size_t)blockIdx.x * blockDim.x + threadIdx.x;
    if (t >= (size_t)BATCH * M3 * 128) return;
    int tt = (int)(t & 127);
    int o  = (int)((t >> 7) % M3);
    int b  = (int)(t / ((size_t)128 * M3));
    float v = 0.f;
    if (tt < 100) v = WB[(size_t)o * 100 + tt] * wa[b * 100 + tt];
    A[t] = f2bf(v);
}

// ---------------- weight-gen stage 2: pack WC^T into WMMA B-fragment swizzle
// wcf index = ((iit*4 + kc)*512 + lane*16 + elem); row K=t, col ii = iit*16 + lane%16
template <int CIN>
__global__ void wcpack_kernel(const float* __restrict__ WC, unsigned short* __restrict__ wcf) {
    constexpr int N3 = 3 * CIN;
    int t = blockIdx.x * blockDim.x + threadIdx.x;
    if (t >= N3 * 128) return;
    int elem = t & 15;
    int lane = (t >> 4) & 31;
    int kc   = (t >> 9) & 3;
    int iit  = t >> 11;
    int tt = kc * 32 + ((lane >> 4) << 4) + elem;    // K row 0..127
    int ii = iit * 16 + (lane & 15);
    float v = (tt < 100) ? WC[(size_t)ii * 100 + tt] : 0.f;
    wcf[t] = f2bf(v);
}

// ---------------- weight-gen stage 3: WMMA GEMM w3[b,o,ii] = A[b] @ WC^T,
// epilogue scatters through the reshape bijection into conv B-fragment layout.
template <int CIN, int COUT>
__global__ void __launch_bounds__(256)
wgemm_kernel(const unsigned short* __restrict__ A,     // [b][3*COUT][128] bf16
             const unsigned short* __restrict__ wcf,   // packed WC fragments
             unsigned short* __restrict__ wfrag) {
    constexpr int M3 = 3 * COUT;
    constexpr int N3 = 3 * CIN;
    constexpr int NM = M3 / 16;
    constexpr int NG = N3 / 64;
    constexpr int NKC  = (CIN * 9) / 32;
    constexpr int NCOT = COUT / 16;
    int wid  = (int)(((size_t)blockIdx.x * blockDim.x + threadIdx.x) >> 5);
    int lane = threadIdx.x & 31;
    int ng = wid % NG;
    int mo = (wid / NG) % NM;
    int b  = wid / (NG * NM);
    int m = lane & 15, hi = lane >> 4;

    v8f acc[4];
    #pragma unroll
    for (int g = 0; g < 4; ++g) acc[g] = (v8f){0.f,0.f,0.f,0.f,0.f,0.f,0.f,0.f};

    const unsigned short* ap = A + ((size_t)(b * M3 + mo * 16 + m)) * 128 + hi * 16;
    #pragma unroll
    for (int kc = 0; kc < 4; ++kc) {
        v8bf alo = *(const v8bf*)(ap + kc * 32);
        v8bf ahi = *(const v8bf*)(ap + kc * 32 + 8);
        v16bf a = __builtin_shufflevector(alo, ahi,
                    0,1,2,3,4,5,6,7,8,9,10,11,12,13,14,15);
        #pragma unroll
        for (int g = 0; g < 4; ++g) {
            v16bf bm = *(const v16bf*)(wcf +
                        (size_t)(((ng * 4 + g) * 4 + kc) * 512 + lane * 16));
            acc[g] = __builtin_amdgcn_wmma_f32_16x16x32_bf16(
                false, a, false, bm, (short)0, acc[g], false, false);
        }
    }

    // scatter: (o,ii) -> co=o/3, q=o%3, f=q*3CIN+ii, ci=f/9, kh,kw=f%9
    //          -> k=(kh*3+kw)*CIN+ci -> fragment (cot,kc,lane,elem)
    #pragma unroll
    for (int g = 0; g < 4; ++g) {
        int ii = (ng * 4 + g) * 16 + m;
        #pragma unroll
        for (int r = 0; r < 8; ++r) {
            int o  = mo * 16 + r + 8 * hi;
            int co = o / 3, q = o % 3;
            int f  = q * N3 + ii;
            int ci = f / 9, rem = f % 9;
            int k  = ((rem / 3) * 3 + (rem % 3)) * CIN + ci;
            int kc2 = k >> 5, kk = k & 31;
            int lw = ((kk >> 4) << 4) + (co & 15);
            int ew = kk & 15;
            size_t idx = (((size_t)(b * NCOT + (co >> 4)) * NKC + kc2) * 512)
                         + lw * 16 + ew;
            wfrag[idx] = f2bf(acc[g][r]);
        }
    }
}

// ---------------- WMMA implicit-GEMM conv (3x3, dil=2, pad=2) ----------------
// block = 8 waves sharing (b, co-group of 64); each wave: 16 pixels x 64 cout.
// B fragments for current (kh,kw) tap staged in LDS via TDM (or coop copy).
template <int CIN, int COUT>
__global__ void __launch_bounds__(256)
conv_wmma_bf16(const unsigned short* __restrict__ xp,
               const unsigned short* __restrict__ wfrag,
               unsigned short* __restrict__ xnext,
               float* __restrict__ stats) {
    constexpr int NKC_C = CIN / 32;        // K-chunks per tap
    constexpr int NKC   = NKC_C * 9;
    constexpr int NCOT  = COUT / 16;
    constexpr int NCOG  = NCOT / 4;        // co-groups of 64
    __shared__ unsigned short smB[4 * NKC_C * 512];

    const int wv   = threadIdx.x >> 5;
    const int lane = threadIdx.x & 31;
    const int tg   = blockIdx.x & 31;                  // 8 pixel tiles per block
    const int cog  = (blockIdx.x >> 5) % NCOG;
    const int b    = (blockIdx.x >> 5) / NCOG;
    const int pt   = tg * 8 + wv;                      // 0..255
    const int y    = pt >> 2;
    const int x0   = (pt & 3) << 4;
    const int m    = lane & 15;
    const int hi   = lane >> 4;

    v8f acc[4];
    #pragma unroll
    for (int g = 0; g < 4; ++g) acc[g] = (v8f){0.f,0.f,0.f,0.f,0.f,0.f,0.f,0.f};

    const size_t wbase = ((size_t)(b * NCOT + cog * 4) * NKC) * 512;  // shorts

    for (int khw = 0; khw < 9; ++khw) {
        __syncthreads();   // previous tap's smB fully consumed
#if USE_TDM
        if (wv == 0) {
            unsigned long long gaddr =
                (unsigned long long)(const void*)(wfrag + wbase + (size_t)khw * NKC_C * 512);
            unsigned lds_off = (unsigned)(unsigned long long)(const void*)smB;
            tdm_load_2d(lds_off, gaddr, NKC_C * 128u, NKC * 128u, 4u);
        }
#else
        {   // cooperative copy fallback: 4 rows of NKC_C*512 shorts
            const unsigned* src =
                (const unsigned*)(wfrag + wbase + (size_t)khw * NKC_C * 512);
            unsigned* dst = (unsigned*)smB;
            constexpr int ROWW = NKC_C * 256;          // u32 per row
            for (int idx = threadIdx.x; idx < 4 * ROWW; idx += 256) {
                int g = idx / ROWW, r = idx % ROWW;
                dst[idx] = src[(size_t)g * NKC * 256 + r];
            }
        }
#endif
        __syncthreads();   // smB ready

        const int yy = y + 2 * (khw / 3);
        const int xx = x0 + m + 2 * (khw % 3);
        const unsigned short* ap =
            xp + ((size_t)((b * PW + yy) * PW + xx)) * CIN + hi * 16;
        const unsigned short* bsm = smB + lane * 16;

        #pragma unroll 2
        for (int cc = 0; cc < NKC_C; ++cc) {
            // A: contiguous k order, same as the B fragments (lane<16: K0-15 ; lane>=16: K16-31)
            v8bf alo = *(const v8bf*)(ap + cc * 32);
            v8bf ahi = *(const v8bf*)(ap + cc * 32 + 8);
            v16bf a = __builtin_shufflevector(alo, ahi,
                        0,1,2,3,4,5,6,7,8,9,10,11,12,13,14,15);
            #pragma unroll
            for (int g = 0; g < 4; ++g) {
                v16bf bm = *(const v16bf*)(bsm + (size_t)(g * NKC_C + cc) * 512);
                acc[g] = __builtin_amdgcn_wmma_f32_16x16x32_bf16(
                    false, a, false, bm, (short)0, acc[g], false, false);
            }
        }
    }

    // epilogue: D layout -> lane column co = m, rows M = r + 8*hi (pixel x = x0+M)
    const int xb = x0 + 8 * hi + 2;
    #pragma unroll
    for (int g = 0; g < 4; ++g) {
        int co = (cog * 4 + g) * 16 + m;
        float s = 0.f, s2 = 0.f;
        size_t row = ((size_t)((b * PW + y + 2) * PW + xb)) * COUT + co;
        #pragma unroll
        for (int r = 0; r < 8; ++r) {
            float v = acc[g][r];
            s += v; s2 += v * v;
            xnext[row + (size_t)r * COUT] = f2bf(v);
        }
        atomicAdd(&stats[co], s);
        atomicAdd(&stats[COUT + co], s2);
    }
}

// ---------------- BN finalize ----------------
__global__ void bn_stats(float* __restrict__ stats, int cout) {
    int c = blockIdx.x * blockDim.x + threadIdx.x;
    if (c >= cout) return;
    float mean = stats[c] * (1.f / BN_COUNT);
    float var  = stats[cout + c] * (1.f / BN_COUNT) - mean * mean;
    stats[2 * cout + c] = mean;
    stats[3 * cout + c] = rsqrtf(var + 1e-5f);
}

// ---------------- BN + ReLU in place (cout is a power of two) ----------------
__global__ void bn_relu(unsigned short* __restrict__ xp, const float* __restrict__ stats,
                        int cout, int cshift) {
    size_t t = (size_t)blockIdx.x * blockDim.x + threadIdx.x;
    if (t >= ((size_t)BATCH * NPIX << cshift)) return;
    int c = (int)(t & (cout - 1));
    size_t p = t >> cshift;
    int xw = (int)(p & 63); p >>= 6;
    int yw = (int)(p & 63);
    int b  = (int)(p >> 6);
    size_t idx = ((size_t)((b * PW + yw + 2) * PW + xw + 2)) * cout + c;
    float v = (bf2f(xp[idx]) - stats[2 * cout + c]) * stats[3 * cout + c];
    xp[idx] = f2bf(v > 0.f ? v : 0.f);
}

// ---------------- 1x1 conv (64->1) + bias + nearest upsample x8 ----------------
__global__ void out_upsample(const unsigned short* __restrict__ xp,
                             const float* __restrict__ ow, const float* __restrict__ ob,
                             float* __restrict__ out) {
    int t = blockIdx.x * blockDim.x + threadIdx.x;
    if (t >= BATCH * NPIX) return;
    int b = t >> 12, p = t & 4095;
    int yy = p >> 6, xx = p & 63;
    const unsigned short* px = xp + ((size_t)((b * PW + yy + 2) * PW + xx + 2)) * 64;
    float s = ob[0];
    #pragma unroll 8
    for (int c = 0; c < 64; ++c) s += bf2f(px[c]) * ow[c];
    float* o = out + (((size_t)b * 512 + (size_t)yy * 8) * 512) + (size_t)xx * 8;
    #pragma unroll
    for (int dy = 0; dy < 8; ++dy)
        #pragma unroll
        for (int dx = 0; dx < 8; ++dx) o[(size_t)dy * 512 + dx] = s;
}

// ---------------- host ----------------
static const int CIN_L[6]  = {512, 512, 512, 512, 256, 128};
static const int COUT_L[6] = {512, 512, 512, 256, 128, 64};
static const int CSH_L[6]  = {9, 9, 9, 8, 7, 6};

extern "C" void kernel_launch(void* const* d_in, const int* in_sizes, int n_in,
                              void* d_out, int out_size, void* d_ws, size_t ws_size,
                              hipStream_t stream) {
    (void)in_sizes; (void)n_in; (void)out_size; (void)ws_size;
    const float* x    = (const float*)d_in[0];
    const float* y    = (const float*)d_in[1];
    const float* fc_w = (const float*)d_in[2];
    const float* fc_b = (const float*)d_in[3];
    const float* WBs[6], *WCs[6];
    for (int i = 0; i < 6; ++i) { WBs[i] = (const float*)d_in[4 + 2 * i];
                                  WCs[i] = (const float*)d_in[5 + 2 * i]; }
    const float* out_w = (const float*)d_in[16];
    const float* out_b = (const float*)d_in[17];
    float* out = (float*)d_out;

    const size_t SXP = (size_t)BATCH * PW * PW * 512 * 2;
    const size_t SWF = (size_t)BATCH * 32 * 144 * 512 * 2;
    const size_t SAS = (size_t)BATCH * 1536 * 128 * 2;      // scaled A
    char* ws = (char*)d_ws;
    unsigned short* xpA   = (unsigned short*)(ws);
    unsigned short* xpB   = (unsigned short*)(ws + SXP);
    unsigned short* wfrag = (unsigned short*)(ws + 2 * SXP);
    float* wa    = (float*)(ws + 2 * SXP + SWF);
    float* stats = (float*)(ws + 2 * SXP + SWF + 4096);
    unsigned short* ascaled = (unsigned short*)(ws + 2 * SXP + SWF + 4096 + 16384);
    unsigned short* wcfrag  = (unsigned short*)(ws + 2 * SXP + SWF + 4096 + 16384 + SAS);

    const int TPB = 256;
    {
        size_t nw = SXP / 4;
        fill_zero_u32<<<(unsigned)((nw + TPB - 1) / TPB), TPB, 0, stream>>>((unsigned*)xpA, nw);
        size_t n = (size_t)BATCH * 512 * NPIX;
        pack_input<<<(unsigned)((n + TPB - 1) / TPB), TPB, 0, stream>>>(x, xpA);
    }

    unsigned short* cur = xpA;
    unsigned short* nxt = xpB;
    for (int i = 0; i < 6; ++i) {
        const int cin = CIN_L[i], cout = COUT_L[i];
        const int ncot = cout >> 4;

        hyper_wa<<<(BATCH * 100 + TPB - 1) / TPB, TPB, 0, stream>>>(
            y, fc_w + (size_t)i * 100 * 512, fc_b + (size_t)i * 100, wa);

        // -------- WMMA weight generation --------
        size_t nsc = (size_t)BATCH * 3 * cout * 128;
        unsigned gsc = (unsigned)((nsc + TPB - 1) / TPB);
        size_t npk = (size_t)3 * cin * 128;
        unsigned gpk = (unsigned)((npk + TPB - 1) / TPB);
        unsigned gge = (unsigned)(BATCH * (3 * cout / 16) * (3 * cin / 64) / 8);
        switch (i) {
            case 0: case 1: case 2:
                wscale_kernel<512><<<gsc, TPB, 0, stream>>>(WBs[i], wa, ascaled);
                wcpack_kernel<512><<<gpk, TPB, 0, stream>>>(WCs[i], wcfrag);
                wgemm_kernel<512, 512><<<gge, TPB, 0, stream>>>(ascaled, wcfrag, wfrag);
                break;
            case 3:
                wscale_kernel<256><<<gsc, TPB, 0, stream>>>(WBs[i], wa, ascaled);
                wcpack_kernel<512><<<gpk, TPB, 0, stream>>>(WCs[i], wcfrag);
                wgemm_kernel<512, 256><<<gge, TPB, 0, stream>>>(ascaled, wcfrag, wfrag);
                break;
            case 4:
                wscale_kernel<128><<<gsc, TPB, 0, stream>>>(WBs[i], wa, ascaled);
                wcpack_kernel<256><<<gpk, TPB, 0, stream>>>(WCs[i], wcfrag);
                wgemm_kernel<256, 128><<<gge, TPB, 0, stream>>>(ascaled, wcfrag, wfrag);
                break;
            case 5:
                wscale_kernel<64><<<gsc, TPB, 0, stream>>>(WBs[i], wa, ascaled);
                wcpack_kernel<128><<<gpk, TPB, 0, stream>>>(WCs[i], wcfrag);
                wgemm_kernel<128, 64><<<gge, TPB, 0, stream>>>(ascaled, wcfrag, wfrag);
                break;
        }

        // -------- conv + BN --------
        size_t nz = ((size_t)BATCH * PW * PW * cout * 2) / 4;
        fill_zero_u32<<<(unsigned)((nz + TPB - 1) / TPB), TPB, 0, stream>>>((unsigned*)nxt, nz);
        fill_zero_u32<<<(2 * cout + TPB - 1) / TPB, TPB, 0, stream>>>(
            (unsigned*)stats, (size_t)2 * cout);

        unsigned blocks = (unsigned)(BATCH * (ncot / 4) * 32);
        switch (i) {
            case 0: case 1: case 2:
                conv_wmma_bf16<512, 512><<<blocks, TPB, 0, stream>>>(cur, wfrag, nxt, stats); break;
            case 3:
                conv_wmma_bf16<512, 256><<<blocks, TPB, 0, stream>>>(cur, wfrag, nxt, stats); break;
            case 4:
                conv_wmma_bf16<256, 128><<<blocks, TPB, 0, stream>>>(cur, wfrag, nxt, stats); break;
            case 5:
                conv_wmma_bf16<128, 64><<<blocks, TPB, 0, stream>>>(cur, wfrag, nxt, stats); break;
        }

        bn_stats<<<(cout + TPB - 1) / TPB, TPB, 0, stream>>>(stats, cout);

        size_t nn = (size_t)BATCH * NPIX * cout;
        bn_relu<<<(unsigned)((nn + TPB - 1) / TPB), TPB, 0, stream>>>(nxt, stats, cout, CSH_L[i]);

        unsigned short* t = cur; cur = nxt; nxt = t;
    }

    out_upsample<<<(BATCH * NPIX + TPB - 1) / TPB, TPB, 0, stream>>>(cur, out_w, out_b, out);
}
